// SelfAttention_27762668601608
// MI455X (gfx1250) — matrix operand order repaired and checked by reading, not compiled, not run on hardware
//
#include <hip/hip_runtime.h>
#include <hip/hip_bf16.h>

typedef __attribute__((ext_vector_type(16))) __bf16 v16bf;
typedef __attribute__((ext_vector_type(8)))  __bf16 v8bf;
typedef __attribute__((ext_vector_type(8)))  float  v8f;

#define NH 16
#define HD 128

// ---------- helpers ----------
__device__ __forceinline__ v16bf ld_afrag(const __bf16* p) {
  // A-matrix 16x32 bf16 fragment, per-lane: elems 0..15 = K=kb..kb+15 (contiguous)
  v8bf lo = *(const v8bf*)p;
  v8bf hi = *(const v8bf*)(p + 8);
  v16bf r;
#pragma unroll
  for (int e = 0; e < 8; ++e) { r[e] = lo[e]; r[e + 8] = hi[e]; }
  return r;
}

__device__ __forceinline__ v8f vzero8() {
  v8f z = {0.f, 0.f, 0.f, 0.f, 0.f, 0.f, 0.f, 0.f};
  return z;
}

// ---------- elementwise fp32 -> bf16 ----------
__global__ void f32_to_bf16(const float* __restrict__ in, __bf16* __restrict__ out, int n) {
  int i = blockIdx.x * blockDim.x + threadIdx.x;
  if (i < n) out[i] = (__bf16)in[i];
}

// ---------- NT GEMM: C[MxN] f32 = A[MxK] bf16 * B[NxK]^T bf16 ----------
// block = 128 threads (4 waves, 2x2), wave tile 32x64, block tile 64x128
__global__ __launch_bounds__(128)
void gemm_nt_bf16(const __bf16* __restrict__ A, const __bf16* __restrict__ B,
                  float* __restrict__ C, int M, int N, int K) {
  const int lane  = threadIdx.x & 31;
  const int wave  = threadIdx.x >> 5;
  const int wm    = wave >> 1, wn = wave & 1;
  const int mb    = blockIdx.x * 64 + wm * 32;
  const int nb    = blockIdx.y * 128 + wn * 64;
  const int lh    = lane & 15;
  const bool hib  = lane >= 16;

  v8f acc[2][4];
#pragma unroll
  for (int i = 0; i < 2; ++i)
#pragma unroll
    for (int j = 0; j < 4; ++j) acc[i][j] = vzero8();

  for (int k0 = 0; k0 < K; k0 += 32) {
    v16bf a[2];
#pragma unroll
    for (int i = 0; i < 2; ++i) {
      const __bf16* pa = A + (size_t)(mb + i * 16 + lh) * K + k0 + (hib ? 16 : 0);
      a[i] = ld_afrag(pa);
    }
#pragma unroll
    for (int j = 0; j < 4; ++j) {
      const __bf16* pb = B + (size_t)(nb + j * 16 + lh) * K + k0 + (hib ? 16 : 0);
      v16bf bv = *(const v16bf*)pb;  // 16 contiguous K values per lane
#pragma unroll
      for (int i = 0; i < 2; ++i)
        acc[i][j] = __builtin_amdgcn_wmma_f32_16x16x32_bf16(
            false, a[i], false, bv, (short)0, acc[i][j], false, false);
    }
  }

#pragma unroll
  for (int i = 0; i < 2; ++i)
#pragma unroll
    for (int j = 0; j < 4; ++j)
#pragma unroll
      for (int v = 0; v < 8; ++v) {
        int row = mb + i * 16 + v + (hib ? 8 : 0);
        int col = nb + j * 16 + lh;
        C[(size_t)row * N + col] = acc[i][j][v];
      }
}

// ---------- RoPE + (B,T,D) -> (b,h,t,d) bf16 ----------
__global__ void rope_to_heads(const float* __restrict__ X, __bf16* __restrict__ Y,
                              int Bsz, int T) {
  long idx = (long)blockIdx.x * blockDim.x + threadIdx.x;
  long total = (long)Bsz * T * NH * (HD / 2);
  if (idx >= total) return;
  int  i = idx % (HD / 2);        // pair/freq index 0..63
  long r = idx / (HD / 2);
  int  h = r % NH;  r /= NH;
  int  t = r % T;   long b = r / T;
  const int D = NH * HD;
  const float* px = X + ((b * (long)T + t) * D + h * HD + 2 * i);
  float x0 = px[0], x1 = px[1];
  // inv_freq = 10000^(-i/64) = exp(-i * ln(10000)/64)
  float ang = (float)t * __expf(-(float)i * 0.14391156831212787f);
  float c = __cosf(ang), s = __sinf(ang);
  float o0 = x0 * c - x1 * s;
  float o1 = x1 * c + x0 * s;
  __bf16* py = Y + ((((long)b * NH + h) * T + t) * HD + 2 * i);
  py[0] = (__bf16)o0;
  py[1] = (__bf16)o1;
}

// ---------- V: (B,T,D) f32 -> (b,h,d,t) bf16 ----------
__global__ void v_transpose(const float* __restrict__ X, __bf16* __restrict__ Y,
                            int Bsz, int T) {
  long idx = (long)blockIdx.x * blockDim.x + threadIdx.x;
  long total = (long)Bsz * T * NH * HD;
  if (idx >= total) return;
  int  d = idx % HD; long r = idx / HD;
  int  h = r % NH;   r /= NH;
  int  t = r % T;    long b = r / T;
  const int D = NH * HD;
  float v = X[(b * (long)T + t) * D + h * HD + d];
  Y[(((long)b * NH + h) * HD + d) * T + t] = (__bf16)v;
}

// ---------- flash attention (causal) ----------
// grid (B*H, T/64); block 128 = 4 waves; each wave owns 16 query rows.
__global__ __launch_bounds__(128)
void flash_attn(const __bf16* __restrict__ Q, const __bf16* __restrict__ Kb,
                const __bf16* __restrict__ Vt, __bf16* __restrict__ O, int T) {
  const int bh   = blockIdx.x;
  const int b    = bh / NH, h = bh % NH;
  const int wave = threadIdx.x >> 5;
  const int lane = threadIdx.x & 31;
  const int lh   = lane & 15;
  const bool hib = lane >= 16;
  const int qr   = blockIdx.y * 64 + wave * 16;
  const float scale = 0.08838834764831845f;  // 1/sqrt(128)

  const __bf16* qp = Q  + (long)bh * T * HD;
  const __bf16* kp = Kb + (long)bh * T * HD;
  const __bf16* vp = Vt + (long)bh * HD * T;

  // Q fragments: 16 rows x 128 features (4 k-chunks of 32)
  v16bf qa[4];
#pragma unroll
  for (int kk = 0; kk < 4; ++kk)
    qa[kk] = ld_afrag(qp + (long)(qr + lh) * HD + kk * 32 + (hib ? 16 : 0));

  v8f o[8];
#pragma unroll
  for (int j = 0; j < 8; ++j) o[j] = vzero8();
  float mrow[8], lrow[8];
#pragma unroll
  for (int v = 0; v < 8; ++v) { mrow[v] = -3.0e38f; lrow[v] = 0.f; }

  __shared__ __bf16 sP[4][16][32];

  const int kend = qr + 16;  // last key needed (exclusive) for this wave's rows
  for (int k0 = 0; k0 < kend; k0 += 32) {
    // S = (Q . K^T) for 32 keys, as two 16x16 fp32 tiles
    v8f S[2];
#pragma unroll
    for (int c = 0; c < 2; ++c) {
      v8f s = vzero8();
#pragma unroll
      for (int kk = 0; kk < 4; ++kk) {
        const __bf16* p = kp + (long)(k0 + c * 16 + lh) * HD + kk * 32 + (hib ? 16 : 0);
        v16bf bv = *(const v16bf*)p;
        s = __builtin_amdgcn_wmma_f32_16x16x32_bf16(false, qa[kk], false, bv,
                                                    (short)0, s, false, false);
      }
      S[c] = s;
    }
    // scale + causal mask (C layout: row = v + 8*hib + qr, col = c*16 + lh + k0)
#pragma unroll
    for (int c = 0; c < 2; ++c) {
      int key = k0 + c * 16 + lh;
#pragma unroll
      for (int v = 0; v < 8; ++v) {
        int trow = qr + v + (hib ? 8 : 0);
        S[c][v] = (key <= trow) ? S[c][v] * scale : -3.0e38f;
      }
    }
    // per-row max across the 16 lanes of each half-wave
    float mloc[8];
#pragma unroll
    for (int v = 0; v < 8; ++v) mloc[v] = fmaxf(S[0][v], S[1][v]);
#pragma unroll
    for (int off = 1; off < 16; off <<= 1)
#pragma unroll
      for (int v = 0; v < 8; ++v)
        mloc[v] = fmaxf(mloc[v], __shfl_xor(mloc[v], off, 32));

    float corr[8];
#pragma unroll
    for (int v = 0; v < 8; ++v) {
      float mn = fmaxf(mrow[v], mloc[v]);
      corr[v] = __expf(mrow[v] - mn);
      mrow[v] = mn;
    }
    // P = exp(S - m); row sums
#pragma unroll
    for (int c = 0; c < 2; ++c)
#pragma unroll
      for (int v = 0; v < 8; ++v) S[c][v] = __expf(S[c][v] - mrow[v]);
    float rs[8];
#pragma unroll
    for (int v = 0; v < 8; ++v) rs[v] = S[0][v] + S[1][v];
#pragma unroll
    for (int off = 1; off < 16; off <<= 1)
#pragma unroll
      for (int v = 0; v < 8; ++v) rs[v] += __shfl_xor(rs[v], off, 32);
#pragma unroll
    for (int v = 0; v < 8; ++v) lrow[v] = lrow[v] * corr[v] + rs[v];

    // C-layout P -> LDS -> A-fragment layout (bf16)
#pragma unroll
    for (int c = 0; c < 2; ++c)
#pragma unroll
      for (int v = 0; v < 8; ++v)
        sP[wave][v + (hib ? 8 : 0)][c * 16 + lh] = (__bf16)S[c][v];
    asm volatile("s_wait_dscnt 0" ::: "memory");  // intra-wave LDS RAW
    v16bf pa = ld_afrag(&sP[wave][lh][hib ? 16 : 0]);

    // rescale O, then O += P @ V  (V^T rows are contiguous in keys)
#pragma unroll
    for (int j = 0; j < 8; ++j)
#pragma unroll
      for (int v = 0; v < 8; ++v) o[j][v] *= corr[v];
#pragma unroll
    for (int j = 0; j < 8; ++j) {
      const __bf16* p = vp + (long)(j * 16 + lh) * T + k0 + (hib ? 16 : 0);
      v16bf bv = *(const v16bf*)p;
      o[j] = __builtin_amdgcn_wmma_f32_16x16x32_bf16(false, pa, false, bv,
                                                     (short)0, o[j], false, false);
    }
  }

  // epilogue: normalize and write bf16 into (B,T,D)
  const int D = NH * HD;
#pragma unroll
  for (int j = 0; j < 8; ++j)
#pragma unroll
    for (int v = 0; v < 8; ++v) {
      int trow = qr + v + (hib ? 8 : 0);
      float val = o[j][v] / lrow[v];
      O[((long)b * T + trow) * D + h * HD + j * 16 + lh] = (__bf16)val;
    }
}

// ---------- launch ----------
extern "C" void kernel_launch(void* const* d_in, const int* in_sizes, int n_in,
                              void* d_out, int out_size, void* d_ws, size_t ws_size,
                              hipStream_t stream) {
  const float* x  = (const float*)d_in[0];
  const float* Wq = (const float*)d_in[1];
  const float* Wk = (const float*)d_in[2];
  const float* Wv = (const float*)d_in[3];
  const float* Wo = (const float*)d_in[4];

  const int Bsz = 2, T = 2048, D = 2048;
  const long MT = (long)Bsz * T;          // 4096 rows
  const long NX = MT * D;                 // activation elems
  const long NW = (long)D * D;            // weight elems

  char* w = (char*)d_ws;
  auto carve = [&](size_t bytes) -> void* {
    void* p = (void*)w;
    w += (bytes + 255) & ~(size_t)255;
    return p;
  };
  __bf16* xb  = (__bf16*)carve(NX * 2);
  __bf16* Wqb = (__bf16*)carve(NW * 2);
  __bf16* Wkb = (__bf16*)carve(NW * 2);
  __bf16* Wvb = (__bf16*)carve(NW * 2);
  __bf16* Wob = (__bf16*)carve(NW * 2);
  float*  Sf  = (float*)carve(NX * 4);    // reused fp32 GEMM output
  __bf16* qb  = (__bf16*)carve(NX * 2);
  __bf16* kb  = (__bf16*)carve(NX * 2);
  __bf16* vt  = (__bf16*)carve(NX * 2);
  __bf16* ob  = (__bf16*)carve(NX * 2);

  // 1) convert inputs to bf16
  f32_to_bf16<<<(int)((NX + 255) / 256), 256, 0, stream>>>(x, xb, (int)NX);
  f32_to_bf16<<<(int)((NW + 255) / 256), 256, 0, stream>>>(Wq, Wqb, (int)NW);
  f32_to_bf16<<<(int)((NW + 255) / 256), 256, 0, stream>>>(Wk, Wkb, (int)NW);
  f32_to_bf16<<<(int)((NW + 255) / 256), 256, 0, stream>>>(Wv, Wvb, (int)NW);
  f32_to_bf16<<<(int)((NW + 255) / 256), 256, 0, stream>>>(Wo, Wob, (int)NW);

  dim3 gg((unsigned)(MT / 64), (unsigned)(D / 128));
  long pairs = NX / 2;

  // 2) Q = x @ Wq^T -> RoPE -> (b,h,t,d) bf16
  gemm_nt_bf16<<<gg, 128, 0, stream>>>(xb, Wqb, Sf, (int)MT, D, D);
  rope_to_heads<<<(int)((pairs + 255) / 256), 256, 0, stream>>>(Sf, qb, Bsz, T);
  // 3) K
  gemm_nt_bf16<<<gg, 128, 0, stream>>>(xb, Wkb, Sf, (int)MT, D, D);
  rope_to_heads<<<(int)((pairs + 255) / 256), 256, 0, stream>>>(Sf, kb, Bsz, T);
  // 4) V -> transposed (b,h,d,t)
  gemm_nt_bf16<<<gg, 128, 0, stream>>>(xb, Wvb, Sf, (int)MT, D, D);
  v_transpose<<<(int)((NX + 255) / 256), 256, 0, stream>>>(Sf, vt, Bsz, T);

  // 5) flash attention -> (B,T,D) bf16
  dim3 ga((unsigned)(Bsz * NH), (unsigned)(T / 64));
  flash_attn<<<ga, 128, 0, stream>>>(qb, kb, vt, ob, T);

  // 6) out = attn @ Wo^T (fp32 to d_out)
  gemm_nt_bf16<<<gg, 128, 0, stream>>>(ob, Wob, (float*)d_out, (int)MT, D, D);
}
